// BitGRUCell_49031346651241
// MI455X (gfx1250) — matrix operand order repaired and checked by reading, not compiled, not run on hardware
//
#include <hip/hip_runtime.h>

// ---------------------------------------------------------------------------
// BitGRU cell for MI455X (gfx1250, wave32, WMMA)
//   B = 4096 batch, I = 2048, H = 2048, fan_in K = 4096
//   sign(w)*mean|w| binarization -> f16 (+/-1) operands, f32 accumulate.
//   Wave tile 64x64 (4x4 WMMA tiles), block tile 128x256 (8 waves).
// ---------------------------------------------------------------------------

typedef __attribute__((ext_vector_type(16))) _Float16 v16h;
typedef __attribute__((ext_vector_type(8)))  _Float16 h8;
typedef __attribute__((ext_vector_type(8)))  float    v8f;

#define BATCH 4096
#define HDIM  2048
#define KDIM  4096                       // fan_in = I + H
#define WELEMS (HDIM * KDIM)             // 8388608 per weight tensor

// ---------------------------------------------------------------------------
// scale init + abs-mean reduction (3 weight tensors)
// ---------------------------------------------------------------------------
__global__ void k_zero3(float* s) {
    if (threadIdx.x < 3) s[threadIdx.x] = 0.0f;
}

__global__ void k_absmean(const float* __restrict__ w0,
                          const float* __restrict__ w1,
                          const float* __restrict__ w2,
                          float* __restrict__ sums) {
    const int which = blockIdx.x >> 8;          // 256 blocks per weight
    const int chunk = blockIdx.x & 255;
    const float* w = (which == 0) ? w0 : (which == 1) ? w1 : w2;
    const int per_chunk = WELEMS / 256;         // 32768
    const size_t start = (size_t)chunk * per_chunk;
    float s = 0.0f;
    for (int i = threadIdx.x; i < per_chunk; i += blockDim.x)
        s += fabsf(w[start + i]);
    __shared__ float red[256];
    red[threadIdx.x] = s;
    __syncthreads();
    for (int off = 128; off > 0; off >>= 1) {
        if (threadIdx.x < off) red[threadIdx.x] += red[threadIdx.x + off];
        __syncthreads();
    }
    if (threadIdx.x == 0) atomicAdd(&sums[which], red[0]);
}

// ---------------------------------------------------------------------------
// binarize weights -> f16 (+/-1), cast activations -> f16
// ---------------------------------------------------------------------------
__global__ void k_sign16(const float* __restrict__ w, _Float16* __restrict__ o, int n) {
    for (int i = blockIdx.x * blockDim.x + threadIdx.x; i < n;
         i += gridDim.x * blockDim.x) {
        float v = w[i];
        float s = (v > 0.0f) ? 1.0f : ((v < 0.0f) ? -1.0f : 0.0f);
        o[i] = (_Float16)s;
    }
}

__global__ void k_cast16(const float* __restrict__ a, _Float16* __restrict__ o, int n) {
    for (int i = blockIdx.x * blockDim.x + threadIdx.x; i < n;
         i += gridDim.x * blockDim.x)
        o[i] = (_Float16)a[i];
}

// ---------------------------------------------------------------------------
// WMMA GEMM: out[row][col] = act( scale * sum_k A[row][k]*Wsign[col][k] + bias[col] )
//   A = [A0 | A1] logically [BATCH, 4096], each half [BATCH, 2048] row-major f16.
//   Wsign row-major [HDIM, 4096] f16 (+/-1).
//   Block = 128x256 (8 waves as 2x4; each wave 64x64 = 4 M-tiles x 4 N-tiles).
//   GATE: 0 = r (writes rh16), 1 = z (writes z16), 2 = n (writes h_new f32)
// ---------------------------------------------------------------------------
template <int GATE>
__global__ __launch_bounds__(256)
void bitgru_gemm(const _Float16* __restrict__ A0,
                 const _Float16* __restrict__ A1,
                 const _Float16* __restrict__ Wsign,
                 const float*    __restrict__ bias,
                 const float*    __restrict__ scale_sum,   // one float (sum |w|)
                 const float*    __restrict__ hidden,      // [BATCH, HDIM] f32
                 const _Float16* __restrict__ zbuf,        // [BATCH, HDIM] f16 (GATE 2)
                 _Float16*       __restrict__ out16,       // rh (GATE 0) / z (GATE 1)
                 float*          __restrict__ outf)        // h_new (GATE 2)
{
    const int lane = threadIdx.x & 31;
    const int wave = threadIdx.x >> 5;
    const int wm   = wave & 1;            // 2 waves along M
    const int wn   = wave >> 1;           // 4 waves along N
    const int m0   = blockIdx.x * 128 + wm * 64;
    const int n0   = blockIdx.y * 256 + wn * 64;
    const int lr   = lane & 15;           // A row-in-tile / B column-in-tile
    const int lh   = lane >> 4;           // half-wave selector

    const float scale = scale_sum[0] * (1.0f / (float)WELEMS);

    v8f acc[4][4];
#pragma unroll
    for (int mi = 0; mi < 4; ++mi)
#pragma unroll
        for (int ni = 0; ni < 4; ++ni)
            acc[mi][ni] = (v8f){0.f, 0.f, 0.f, 0.f, 0.f, 0.f, 0.f, 0.f};

    // row-base pointers (constant across the K loop)
    const _Float16* arow[4];
#pragma unroll
    for (int mi = 0; mi < 4; ++mi)
        arow[mi] = (const _Float16*)((size_t)(m0 + mi * 16 + lr) * 2048 * sizeof(_Float16));
    const _Float16* wrow[4];
#pragma unroll
    for (int ni = 0; ni < 4; ++ni)
        wrow[ni] = Wsign + (size_t)(n0 + ni * 16 + lr) * (size_t)KDIM;

    for (int half = 0; half < 2; ++half) {
        const _Float16* Abase = (half == 0) ? A0 : A1;
        const int kw0 = half * 2048;      // W column offset for this half

        for (int kk = 0; kk < 2048; kk += 32) {
            // ---- A fragments: lane = row; K chunks {ka..ka+7, ka+16..ka+23}
            const int ka = kk + lh * 8;
            v16h a[4];
#pragma unroll
            for (int mi = 0; mi < 4; ++mi) {
                const _Float16* p = Abase + (size_t)arow[mi] / sizeof(_Float16) + ka;
                h8 lo = *(const h8*)(p);
                h8 hi = *(const h8*)(p + 16);
                v16h av;
#pragma unroll
                for (int i = 0; i < 8; ++i) { av[i] = lo[i]; av[i + 8] = hi[i]; }
                a[mi] = av;
            }

            // ---- B fragments: lane = column; K chunks {kb..kb+7, kb+16..kb+23}
            const int kb = kw0 + kk + lh * 8;
            v16h b[4];
#pragma unroll
            for (int ni = 0; ni < 4; ++ni) {
                const _Float16* p = wrow[ni] + kb;
                h8 lo = *(const h8*)(p);
                h8 hi = *(const h8*)(p + 16);
                v16h bv;
#pragma unroll
                for (int i = 0; i < 8; ++i) { bv[i] = lo[i]; bv[i + 8] = hi[i]; }
                b[ni] = bv;
                __builtin_prefetch(p + 32, 0, 3);   // 64B ahead, keep near
            }

            // ---- 16 WMMAs (64x64 output per wave, K step 32)
#pragma unroll
            for (int mi = 0; mi < 4; ++mi)
#pragma unroll
                for (int ni = 0; ni < 4; ++ni)
                    acc[mi][ni] = __builtin_amdgcn_wmma_f32_16x16x32_f16(
                        /*neg_a=*/false, a[mi], /*neg_b=*/false, b[ni],
                        /*c_mod=*/(short)0, acc[mi][ni],
                        /*reuse_a=*/false, /*reuse_b=*/false);
        }
    }

    // ---- fused epilogue
#pragma unroll
    for (int mi = 0; mi < 4; ++mi) {
#pragma unroll
        for (int ni = 0; ni < 4; ++ni) {
            const int col  = n0 + ni * 16 + lr;
            const float bc = bias[col];
#pragma unroll
            for (int r = 0; r < 8; ++r) {
                const int row  = m0 + mi * 16 + lh * 8 + r;
                const size_t i = (size_t)row * HDIM + col;
                const float v  = scale * acc[mi][ni][r] + bc;
                if (GATE == 0) {
                    const float rg = 1.0f / (1.0f + __expf(-v));
                    out16[i] = (_Float16)(rg * hidden[i]);        // rh for gate n
                } else if (GATE == 1) {
                    out16[i] = (_Float16)(1.0f / (1.0f + __expf(-v)));  // z
                } else {
                    const float n = tanhf(v);
                    const float z = (float)zbuf[i];
                    const float h = hidden[i];
                    outf[i] = (1.0f - z) * h + z * n;             // h_new
                }
            }
        }
    }
}

// ---------------------------------------------------------------------------
// launch
// ---------------------------------------------------------------------------
extern "C" void kernel_launch(void* const* d_in, const int* in_sizes, int n_in,
                              void* d_out, int out_size, void* d_ws, size_t ws_size,
                              hipStream_t stream) {
    const float* x      = (const float*)d_in[0];
    const float* hidden = (const float*)d_in[1];
    const float* w_r    = (const float*)d_in[2];
    const float* b_r    = (const float*)d_in[3];
    const float* w_z    = (const float*)d_in[4];
    const float* b_z    = (const float*)d_in[5];
    const float* w_n    = (const float*)d_in[6];
    const float* b_n    = (const float*)d_in[7];
    float* out = (float*)d_out;

    // workspace layout (all 256B aligned): scales | x16 | h16 | rh16 | z16 | wr16 | wz16 | wn16
    char* ws = (char*)d_ws;
    const size_t ACT = (size_t)BATCH * HDIM * sizeof(_Float16);  // 16 MB
    const size_t WSZ = (size_t)WELEMS * sizeof(_Float16);        // 16 MB
    float*    scales = (float*)ws;
    _Float16* x16  = (_Float16*)(ws + 256);
    _Float16* h16  = (_Float16*)(ws + 256 + ACT);
    _Float16* rh16 = (_Float16*)(ws + 256 + 2 * ACT);
    _Float16* z16  = (_Float16*)(ws + 256 + 3 * ACT);
    _Float16* wr16 = (_Float16*)(ws + 256 + 4 * ACT);
    _Float16* wz16 = (_Float16*)(ws + 256 + 4 * ACT + WSZ);
    _Float16* wn16 = (_Float16*)(ws + 256 + 4 * ACT + 2 * WSZ);

    // 1) per-tensor abs-mean scales
    k_zero3<<<1, 32, 0, stream>>>(scales);
    k_absmean<<<768, 256, 0, stream>>>(w_r, w_z, w_n, scales);

    // 2) binarize weights + cast activations to f16
    k_sign16<<<2048, 256, 0, stream>>>(w_r, wr16, WELEMS);
    k_sign16<<<2048, 256, 0, stream>>>(w_z, wz16, WELEMS);
    k_sign16<<<2048, 256, 0, stream>>>(w_n, wn16, WELEMS);
    k_cast16<<<2048, 256, 0, stream>>>(x,      x16, BATCH * HDIM);
    k_cast16<<<2048, 256, 0, stream>>>(hidden, h16, BATCH * HDIM);

    // 3) three fused WMMA GEMMs
    dim3 grid(BATCH / 128, HDIM / 256);
    bitgru_gemm<0><<<grid, 256, 0, stream>>>(x16, h16,  wr16, b_r, &scales[0],
                                             hidden, nullptr, rh16, nullptr);
    bitgru_gemm<1><<<grid, 256, 0, stream>>>(x16, h16,  wz16, b_z, &scales[1],
                                             hidden, nullptr, z16,  nullptr);
    bitgru_gemm<2><<<grid, 256, 0, stream>>>(x16, rh16, wn16, b_n, &scales[2],
                                             hidden, z16,     nullptr, out);
}
